// AttentionPool_40278203302590
// MI455X (gfx1250) — matrix operand order repaired and checked by reading, not compiled, not run on hardware
//
#include <hip/hip_runtime.h>
#include <hip/hip_bf16.h>
#include <math.h>

#define B_ 32
#define N_ 2048
#define D_ 1024
#define H_ 4
#define F_ 512
#define KSTEPS (D_ / 32)    // 32 k-steps of 32
#define FTILES (F_ / 16)    // 32 f-tiles of 16
#define ROWS_PER_BLK 64     // 4 m-tiles of 16 rows
#define NTILES64 (N_ / ROWS_PER_BLK)   // 32
#define BSTAGE (FTILES * 32 * 16)      // 16384 bf16 = 32 KB per k-step

typedef __attribute__((ext_vector_type(16))) __bf16 v16bf;
typedef __attribute__((ext_vector_type(8)))  float  v8f;
typedef __attribute__((ext_vector_type(4)))  int    v4i;

union Frag {
  uint4 q[2];
  v16bf v;
};

#if defined(__gfx1250__) && __has_builtin(__builtin_amdgcn_global_load_async_to_lds_b128)
#define HAS_ASYNC_LDS 1
#else
#define HAS_ASYNC_LDS 0
#endif

static __device__ __forceinline__ void wait_async0() {
#if HAS_ASYNC_LDS
#if __has_builtin(__builtin_amdgcn_s_wait_asynccnt)
  __builtin_amdgcn_s_wait_asynccnt(0);
#else
  asm volatile("s_wait_asynccnt 0x0" ::: "memory");
#endif
#endif
}

static __device__ __forceinline__ void copy_b128_to_lds(const uint4* g, uint4* l) {
#if HAS_ASYNC_LDS
  typedef __attribute__((address_space(1))) v4i gv4_t;   // global
  typedef __attribute__((address_space(3))) v4i lv4_t;   // LDS
  __builtin_amdgcn_global_load_async_to_lds_b128((gv4_t*)g, (lv4_t*)l, 0, 0);
#else
  *l = *g;
#endif
}

static __device__ __forceinline__ unsigned short f2bf(float f) {
  union { float f; unsigned u; } c;
  c.f = f;
  unsigned u = c.u;
  u += 0x7FFFu + ((u >> 16) & 1u);   // round-to-nearest-even
  return (unsigned short)(u >> 16);
}

// ---------------------------------------------------------------------------
// Pre-swizzle W1 (fp32 [H][D][F]) into bf16 WMMA B-fragment layout:
//   W1bf[h][kt][ft][lane][e] , e = 0..15
//   lane < 16 : f = ft*16 + lane      , K = kt*32 + e
//   lane >=16 : f = ft*16 + (lane-16) , K = kt*32 + 16 + e
// One (h,kt) slab is a contiguous 32 KB block -> ideal for LDS staging.
// ---------------------------------------------------------------------------
__global__ __launch_bounds__(256) void prep_w1(const float* __restrict__ W1,
                                               unsigned short* __restrict__ W1bf) {
  int idx = blockIdx.x * blockDim.x + threadIdx.x;   // 2,097,152 total
  int e    = idx & 15;
  int lane = (idx >> 4) & 31;
  int ft   = (idx >> 9) & 31;
  int kt   = (idx >> 14) & 31;
  int h    = idx >> 19;
  int k = kt * 32 + ((lane >> 4) << 4) + e;
  int f = ft * 16 + (lane & 15);
  W1bf[idx] = f2bf(W1[((size_t)h * D_ + k) * F_ + f]);
}

// ---------------------------------------------------------------------------
// Fused score MLP: scores[b,h,n] = W2[h] . tanh(x[b,n,:] @ W1[h] + b1[h]) + b2[h]
// Block: 512 threads = 16 waves = 4 m-tiles x 4 f-quads -> 64 rows, full F=512.
// B (W1bf) staged once per k-step into LDS (async, double-buffered) and shared
// by all 16 waves; A (x tile) converted fp32->bf16 and staged likewise.
// Inner t-loop is software-pipelined: B fragment t+1 loads while WMMA t runs.
// ---------------------------------------------------------------------------
__global__ __launch_bounds__(512) void attn_scores(
    const float* __restrict__ x, const int* __restrict__ vmask,
    const unsigned short* __restrict__ W1bf,
    const float* __restrict__ b1, const float* __restrict__ W2,
    const float* __restrict__ b2, float* __restrict__ scores)
{
  __shared__ unsigned short As[2][ROWS_PER_BLK * 40]; // 64 rows x 32 k, stride 40 (16B-aligned rows)
  __shared__ unsigned short Bs[2][BSTAGE];            // 2 x 32 KB, fragment layout [ft][lane][16]
  __shared__ float red[ROWS_PER_BLK];

  int tid   = threadIdx.x;
  int ntile = blockIdx.x % NTILES64;
  int bh    = blockIdx.x / NTILES64;
  int h     = bh % H_;
  int b     = bh / H_;
  int n0    = ntile * ROWS_PER_BLK;

  if (tid < ROWS_PER_BLK) red[tid] = 0.f;

  int wave   = tid >> 5;
  int lane   = tid & 31;
  int mtile  = wave & 3;        // 0..3 -> rows [mtile*16, +16)
  int fquad  = wave >> 2;       // 0..3 -> f    [fquad*128, +128)
  int laneLo = lane & 15;
  int laneHi = lane >> 4;

  v8f acc[8];
  #pragma unroll
  for (int t = 0; t < 8; ++t)
    #pragma unroll
    for (int e = 0; e < 8; ++e) acc[t][e] = 0.f;

  // cooperative A-tile load: thread -> (row, 4 consecutive k)
  int arow = tid >> 3;                  // 0..63
  int akq  = (tid & 7) << 2;            // 0,4,..,28
  const float* xrow = x + ((size_t)b * N_ + n0 + arow) * D_;

  // B stage: contiguous 32 KB slab per (h,kt); each thread copies 4 x 16B
  const unsigned short* wslab0 = W1bf + (size_t)h * KSTEPS * BSTAGE;

  // ---- prologue: stage k-step 0 into buffer 0 ----
  {
    float4 xv = *(const float4*)(xrow + akq);
    unsigned a01 = (unsigned)f2bf(xv.x) | ((unsigned)f2bf(xv.y) << 16);
    unsigned a23 = (unsigned)f2bf(xv.z) | ((unsigned)f2bf(xv.w) << 16);
    *(uint2*)&As[0][arow * 40 + akq] = make_uint2(a01, a23);
    const uint4* gs = (const uint4*)wslab0 + tid * 4;
    uint4* ls = (uint4*)&Bs[0][0] + tid * 4;
    #pragma unroll
    for (int c = 0; c < 4; ++c) copy_b128_to_lds(gs + c, ls + c);
  }

  for (int kt = 0; kt < KSTEPS; ++kt) {
    int buf = kt & 1;
    int nxt = buf ^ 1;

    wait_async0();        // this wave's stage-kt async copies done
    __syncthreads();      // all waves' stage-kt writes visible; buffer nxt free

    // ---- prefetch stage kt+1 into the other buffer ----
    if (kt + 1 < KSTEPS) {
      float4 xv = *(const float4*)(xrow + (kt + 1) * 32 + akq);
      unsigned a01 = (unsigned)f2bf(xv.x) | ((unsigned)f2bf(xv.y) << 16);
      unsigned a23 = (unsigned)f2bf(xv.z) | ((unsigned)f2bf(xv.w) << 16);
      *(uint2*)&As[nxt][arow * 40 + akq] = make_uint2(a01, a23);
      const uint4* gs = (const uint4*)(wslab0 + (size_t)(kt + 1) * BSTAGE) + tid * 4;
      uint4* ls = (uint4*)&Bs[nxt][0] + tid * 4;
      #pragma unroll
      for (int c = 0; c < 4; ++c) copy_b128_to_lds(gs + c, ls + c);
    }

    // ---- consume stage kt ----
    // A fragment: K runs [hi*16, hi*16+16)
    Frag af;
    const uint4* ap = (const uint4*)&As[buf][(mtile * 16 + laneLo) * 40 + laneHi * 16];
    af.q[0] = ap[0];
    af.q[1] = ap[1];

    // software-pipelined over the 8 f-tiles: load t+1 while WMMA t executes
    Frag bcur;
    {
      const uint4* bp = (const uint4*)&Bs[buf][((fquad * 8) * 32 + lane) * 16];
      bcur.q[0] = bp[0];
      bcur.q[1] = bp[1];
    }
    #pragma unroll
    for (int t = 0; t < 8; ++t) {
      Frag bnext;
      if (t + 1 < 8) {
        const uint4* bp = (const uint4*)&Bs[buf][((fquad * 8 + t + 1) * 32 + lane) * 16];
        bnext.q[0] = bp[0];
        bnext.q[1] = bp[1];
      }
      acc[t] = __builtin_amdgcn_wmma_f32_16x16x32_bf16(
          false, af.v, false, bcur.v, (short)0, acc[t], false, false);
      if (t + 1 < 8) bcur = bnext;
    }
  }

  // ---- epilogue: + b1, tanh, * W2, reduce over F ----
  float partial[8];
  #pragma unroll
  for (int j = 0; j < 8; ++j) partial[j] = 0.f;

  #pragma unroll
  for (int t = 0; t < 8; ++t) {
    int f = fquad * 128 + t * 16 + laneLo;
    float bb = b1[h * F_ + f];
    float ww = W2[h * F_ + f];
    #pragma unroll
    for (int j = 0; j < 8; ++j)
      partial[j] += tanhf(acc[t][j] + bb) * ww;
  }

  #pragma unroll
  for (int j = 0; j < 8; ++j) {
    float v = partial[j];
    v += __shfl_xor(v, 1);
    v += __shfl_xor(v, 2);
    v += __shfl_xor(v, 4);
    v += __shfl_xor(v, 8);
    if (laneLo == 0) {
      int m = mtile * 16 + laneHi * 8 + j;   // C layout: lanes>=16 -> M+8
      atomicAdd(&red[m], v);
    }
  }
  __syncthreads();

  if (tid < ROWS_PER_BLK) {
    int n = n0 + tid;
    float s = red[tid] + b2[h];
    if (vmask[b * N_ + n] == 0) s = -1e9f;
    scores[((size_t)b * H_ + h) * N_ + n] = s;
  }
}

// ---------------------------------------------------------------------------
// Softmax over N per (b,h); in place.
// ---------------------------------------------------------------------------
__global__ __launch_bounds__(256) void softmax_n(float* __restrict__ scores) {
  __shared__ float sred[256];
  float* s = scores + (size_t)blockIdx.x * N_;
  int tid = threadIdx.x;

  float vals[8];
  float m = -INFINITY;
  #pragma unroll
  for (int i = 0; i < 8; ++i) { vals[i] = s[tid + i * 256]; m = fmaxf(m, vals[i]); }
  sred[tid] = m;
  __syncthreads();
  for (int off = 128; off > 0; off >>= 1) {
    if (tid < off) sred[tid] = fmaxf(sred[tid], sred[tid + off]);
    __syncthreads();
  }
  m = sred[0];
  __syncthreads();

  float sum = 0.f;
  #pragma unroll
  for (int i = 0; i < 8; ++i) { vals[i] = __expf(vals[i] - m); sum += vals[i]; }
  sred[tid] = sum;
  __syncthreads();
  for (int off = 128; off > 0; off >>= 1) {
    if (tid < off) sred[tid] += sred[tid + off];
    __syncthreads();
  }
  float inv = 1.f / sred[0];
  #pragma unroll
  for (int i = 0; i < 8; ++i) s[tid + i * 256] = vals[i] * inv;
}

__global__ void zero_f32(float* __restrict__ p, int n) {
  int i = blockIdx.x * blockDim.x + threadIdx.x;
  if (i < n) p[i] = 0.f;
}

// ---------------------------------------------------------------------------
// pooled[b,h,d] = sum_n w[b,h,n] * x[b,n,d]   (second streaming pass over x)
// grid (B, 8): each block handles 256 n's, full D via float4/thread.
// ---------------------------------------------------------------------------
__global__ __launch_bounds__(256) void pool_x(
    const float* __restrict__ x, const float* __restrict__ w,
    float* __restrict__ pooled)
{
  __shared__ float wl[H_ * 256];
  int b  = blockIdx.x;
  int n0 = blockIdx.y * 256;
  int tid = threadIdx.x;

  #pragma unroll
  for (int h = 0; h < H_; ++h)
    wl[h * 256 + tid] = w[((size_t)b * H_ + h) * N_ + n0 + tid];
  __syncthreads();

  float acc[H_][4];
  #pragma unroll
  for (int h = 0; h < H_; ++h)
    #pragma unroll
    for (int c = 0; c < 4; ++c) acc[h][c] = 0.f;

  const float* xb = x + ((size_t)b * N_ + n0) * D_ + tid * 4;
  for (int n = 0; n < 256; ++n) {
    float4 xv = *(const float4*)(xb + (size_t)n * D_);
    #pragma unroll
    for (int h = 0; h < H_; ++h) {
      float ww = wl[h * 256 + n];
      acc[h][0] += ww * xv.x;
      acc[h][1] += ww * xv.y;
      acc[h][2] += ww * xv.z;
      acc[h][3] += ww * xv.w;
    }
  }
  #pragma unroll
  for (int h = 0; h < H_; ++h)
    #pragma unroll
    for (int c = 0; c < 4; ++c)
      atomicAdd(&pooled[((size_t)b * H_ + h) * D_ + tid * 4 + c], acc[h][c]);
}

// ---------------------------------------------------------------------------
// out[b,:] = pooled[b,:] @ Wout + bout      ([32 x 4096] x [4096 x 1024])
// ---------------------------------------------------------------------------
__global__ __launch_bounds__(256) void out_proj(
    const float* __restrict__ pooled, const float* __restrict__ Wout,
    const float* __restrict__ bout, float* __restrict__ out)
{
  __shared__ float pl[H_ * D_];   // 16 KB
  int b = blockIdx.x;
  int d = blockIdx.y * 256 + threadIdx.x;

  for (int i = threadIdx.x; i < H_ * D_; i += 256)
    pl[i] = pooled[(size_t)b * H_ * D_ + i];
  __syncthreads();

  float acc = bout[d];
  #pragma unroll 8
  for (int k = 0; k < H_ * D_; ++k)
    acc += pl[k] * Wout[(size_t)k * D_ + d];
  out[(size_t)b * D_ + d] = acc;
}

// ---------------------------------------------------------------------------
extern "C" void kernel_launch(void* const* d_in, const int* in_sizes, int n_in,
                              void* d_out, int out_size, void* d_ws, size_t ws_size,
                              hipStream_t stream)
{
  const float* x    = (const float*)d_in[0];
  const int*   vm   = (const int*)  d_in[1];
  const float* W1   = (const float*)d_in[2];
  const float* b1   = (const float*)d_in[3];
  const float* W2   = (const float*)d_in[4];
  const float* b2   = (const float*)d_in[5];
  const float* Wout = (const float*)d_in[6];
  const float* bout = (const float*)d_in[7];
  float* out = (float*)d_out;

  char* ws = (char*)d_ws;
  unsigned short* W1bf  = (unsigned short*)ws;                       // 4 MB
  float*          score = (float*)(ws + (4u << 20));                 // 1 MB
  float*          pool  = (float*)(ws + (4u << 20) + (1u << 20));    // 512 KB

  prep_w1   <<<(H_ * KSTEPS * FTILES * 32 * 16) / 256, 256, 0, stream>>>(W1, W1bf);
  attn_scores<<<B_ * H_ * NTILES64, 512, 0, stream>>>(x, vm, W1bf, b1, W2, b2, score);
  softmax_n <<<B_ * H_, 256, 0, stream>>>(score);
  zero_f32  <<<(B_ * H_ * D_) / 256, 256, 0, stream>>>(pool, B_ * H_ * D_);
  pool_x    <<<dim3(B_, 8), 256, 0, stream>>>(x, score, pool);
  out_proj  <<<dim3(B_, 4), 256, 0, stream>>>(pool, Wout, bout, out);
}
